// NonLocalBlock2D_46626164965556
// MI455X (gfx1250) — matrix operand order repaired and checked by reading, not compiled, not run on hardware
//
#include <hip/hip_runtime.h>

// ---------------------------------------------------------------------------
// Non-local block (spatial self-attention), MI455X / gfx1250, wave32 WMMA.
//   B=8, C=256, CI=64, N=H*W=4096.
// Flash-attention style: the [N,N] score matrix is never materialized in HBM.
// v2: attention workgroup = 8 waves sharing key/value chunks staged in LDS by
// the Tensor Data Mover (double-buffered, TENSORcnt-tracked) -> 8x less L2
// traffic and DMA/compute overlap. Fallback: cooperative global->LDS copies.
// ---------------------------------------------------------------------------

#define NB  8
#define NC  256
#define NCI 64
#define NN  4096

#define QTILE  128   // query rows per workgroup (8 waves x 16)
#define KCHUNK 64    // key columns staged in LDS per outer step

typedef __attribute__((ext_vector_type(16))) _Float16 v16h;
typedef __attribute__((ext_vector_type(8)))  _Float16 v8h;
typedef __attribute__((ext_vector_type(8)))  float    v8f;
typedef __attribute__((ext_vector_type(4)))  float    v4f;

typedef unsigned int       u32;
typedef unsigned long long u64;
typedef __attribute__((ext_vector_type(4))) u32 v4u;
typedef __attribute__((ext_vector_type(8))) int v8i;
typedef __attribute__((ext_vector_type(4))) int v4i;

// Tensor Data Mover availability (device pass only; host pass -> fallback).
#ifndef USE_TDM
#  if defined(__has_builtin)
#    if __has_builtin(__builtin_amdgcn_tensor_load_to_lds) && \
        __has_builtin(__builtin_amdgcn_s_wait_tensorcnt)
#      define USE_TDM 1
#    else
#      define USE_TDM 0
#    endif
#  else
#    define USE_TDM 0
#  endif
#endif

static __device__ __forceinline__ v8f wmma_f16(v16h a, v16h b, v8f c) {
  // D = A(16x32 f16) * B(32x16 f16) + C(16x16 f32)
  return __builtin_amdgcn_wmma_f32_16x16x32_f16(false, a, false, b,
                                                (short)0, c, false, false);
}

#if USE_TDM
// Issue a TDM load of a 2-D f16 tile (tile_d0 contiguous elems x tile_d1
// lines, line stride = stride0 elems) from global to LDS. D# bitfields per
// CDNA5 ISA 8.3/8.4: group0 = {count=1, lds_addr, global_addr, type=2},
// group1 = {data_size=2B, tensor dims, tile dims, dim0 stride}.
static __device__ __forceinline__ void tdm_load_2d(
    u32 lds_off, const _Float16* gsrc,
    u32 tensor_d0, u32 tensor_d1, u64 stride0, u32 tile_d0, u32 tile_d1)
{
  const u64 ga = (u64)(size_t)gsrc;
  v4u g0;
  g0[0] = 1u;                                              // count=1 (valid)
  g0[1] = lds_off;                                         // lds_addr
  g0[2] = (u32)ga;                                         // global_addr lo
  g0[3] = (u32)((ga >> 32) & 0x01FFFFFFu) | (2u << 30);    // addr hi | type=2

  const u64 q0 = ((u64)1 << 16)                            // data_size=1 (2B)
               | ((u64)(tensor_d0 & 0xFFFFu) << 48);       // tensor_dim0 lo16
  const u64 q1 = (u64)(tensor_d0 >> 16)                    // tensor_dim0 hi16
               | ((u64)tensor_d1 << 16)                    // tensor_dim1
               | ((u64)tile_d0 << 48);                     // tile_dim0
  const u64 q2 = (u64)tile_d1                              // tile_dim1
               | ((u64)(stride0 & 0xFFFFFFFFull) << 32);   // dim0_stride lo32
  const u64 q3 = (u64)((stride0 >> 32) & 0xFFFFull);       // dim0_stride hi16

  v8i g1;
  g1[0] = (int)(u32)q0; g1[1] = (int)(u32)(q0 >> 32);
  g1[2] = (int)(u32)q1; g1[3] = (int)(u32)(q1 >> 32);
  g1[4] = (int)(u32)q2; g1[5] = (int)(u32)(q2 >> 32);
  g1[6] = (int)(u32)q3; g1[7] = (int)(u32)(q3 >> 32);

  v4i z4 = {};
#if __clang_major__ >= 23
  v8i z8 = {};
  __builtin_amdgcn_tensor_load_to_lds(g0, g1, z4, z4, z8, 0);
#else
  __builtin_amdgcn_tensor_load_to_lds(g0, g1, z4, z4, 0);
#endif
}
#endif  // USE_TDM

// ---------------------------------------------------------------------------
// Kernel 1: fused 1x1-conv projections as WMMA GEMMs.
//   theta/phi -> [B][N][CI] f16, g -> [B][CI][N] f16.
// ---------------------------------------------------------------------------
__global__ __launch_bounds__(32) void proj_kernel(
    const float* __restrict__ x,                     // [B][C][N] f32
    const float* __restrict__ th_w, const float* __restrict__ th_b,
    const float* __restrict__ ph_w, const float* __restrict__ ph_b,
    const float* __restrict__ g_w,  const float* __restrict__ g_b,
    _Float16* __restrict__ thetaO,                   // [B][N][CI]
    _Float16* __restrict__ phiO,                     // [B][N][CI]
    _Float16* __restrict__ gO)                       // [B][CI][N]
{
  const int lane = threadIdx.x;
  const int lo = lane & 15, hi = lane >> 4;
  const int rt   = blockIdx.x;       // 0..11 : 12 row-tiles of Wcat[192,256]
  const int proj = rt >> 2;          // 0=theta 1=phi 2=g
  const int o0   = (rt & 3) << 4;
  const int n0   = blockIdx.y << 4;
  const int b    = blockIdx.z;

  const float* W  = proj == 0 ? th_w : (proj == 1 ? ph_w : g_w);
  const float* Bb = proj == 0 ? th_b : (proj == 1 ? ph_b : g_b);

  const float* wrow = W + (size_t)(o0 + lo) * NC;
  const float* xcol = x + (size_t)b * NC * NN + n0 + lo;

  v8f acc = {};
  for (int c0 = 0; c0 < NC; c0 += 32) {
    v4f w0 = *(const v4f*)(wrow + c0 + hi * 8);
    v4f w1 = *(const v4f*)(wrow + c0 + hi * 8 + 4);
    v4f w2 = *(const v4f*)(wrow + c0 + 16 + hi * 8);
    v4f w3 = *(const v4f*)(wrow + c0 + 16 + hi * 8 + 4);
    v16h a;
#pragma unroll
    for (int j = 0; j < 4; ++j) {
      a[j]      = (_Float16)w0[j];
      a[4 + j]  = (_Float16)w1[j];
      a[8 + j]  = (_Float16)w2[j];
      a[12 + j] = (_Float16)w3[j];
    }
    v16h bm;
#pragma unroll
    for (int j = 0; j < 16; ++j)
      bm[j] = (_Float16)xcol[(size_t)(c0 + hi * 8 + (j & 7) + (j >> 3) * 16) * NN];
    acc = wmma_f16(a, bm, acc);
  }

  const int n = n0 + lo;
  if (proj < 2) {
    _Float16* out = (proj == 0 ? thetaO : phiO) +
                    ((size_t)b * NN + n) * NCI + o0 + hi * 8;
    v8h v;
#pragma unroll
    for (int r = 0; r < 8; ++r)
      v[r] = (_Float16)(acc[r] + Bb[o0 + hi * 8 + r]);
    *(v8h*)out = v;
  } else {
#pragma unroll
    for (int r = 0; r < 8; ++r) {
      const int row = o0 + hi * 8 + r;
      gO[((size_t)b * NCI + row) * NN + n] = (_Float16)(acc[r] + Bb[row]);
    }
  }
}

// ---------------------------------------------------------------------------
// Kernel 2: flash attention, 8 waves x 16 query rows per workgroup.
// Key/value chunks (64 cols) staged in LDS: TDM double-buffered when
// available, else cooperative copies. Online softmax in registers.
// ---------------------------------------------------------------------------
__global__ __launch_bounds__(256) void attn_kernel(
    const _Float16* __restrict__ theta,  // [B][N][CI]
    const _Float16* __restrict__ phi,    // [B][N][CI]
    const _Float16* __restrict__ g,      // [B][CI][N]
    _Float16* __restrict__ y)            // [B][N][CI]
{
  __shared__ __align__(32) _Float16 sphi[2][KCHUNK * NCI];  // [key row][ci]
  __shared__ __align__(32) _Float16 sg[2][NCI * KCHUNK];    // [ci][key col]
  __shared__ __align__(32) _Float16 spb[8][16 * 32];        // per-wave P tile

  const int tid  = threadIdx.x;
  const int wave = tid >> 5;
  const int lane = tid & 31;
  const int lo = lane & 15, hi = lane >> 4;
  const int r0 = blockIdx.x * QTILE + wave * 16;
  const int b  = blockIdx.y;

  const _Float16* phib = phi + (size_t)b * NN * NCI;
  const _Float16* gb   = g + (size_t)b * NCI * NN;

  // theta A-tiles (K-chunks of 32 covering CI=64), direct from global.
  const _Float16* trow = theta + ((size_t)b * NN + r0 + lo) * NCI;
  v16h ta0, ta1;
  {
    v8h t0 = *(const v8h*)(trow + hi * 16);
    v8h t1 = *(const v8h*)(trow + hi * 16 + 8);
    v8h t2 = *(const v8h*)(trow + 32 + hi * 16);
    v8h t3 = *(const v8h*)(trow + 40 + hi * 16);
#pragma unroll
    for (int j = 0; j < 8; ++j) {
      ta0[j] = t0[j]; ta0[8 + j] = t1[j];
      ta1[j] = t2[j]; ta1[8 + j] = t3[j];
    }
  }

  float m[8], l[8];
  v8f o0v = {}, o1v = {}, o2v = {}, o3v = {};
#pragma unroll
  for (int r = 0; r < 8; ++r) { m[r] = -1e30f; l[r] = 0.f; }

  const int NK = NN / KCHUNK;  // 64 outer chunks

#if USE_TDM
  // Prologue: DMA chunk 0 into buffer 0 (phi tile: 64 lines x 64 contiguous
  // ci; g tile: 64 lines (ci) x 64 contiguous cols, line stride N).
  if (wave == 0) {
    tdm_load_2d((u32)(size_t)&sphi[0][0], phib, NCI, NN, NCI, NCI, KCHUNK);
    tdm_load_2d((u32)(size_t)&sg[0][0],   gb,   NN, NCI, NN, KCHUNK, NCI);
    __builtin_amdgcn_s_wait_tensorcnt(0);
  }
  __syncthreads();
#endif

  for (int k = 0; k < NK; ++k) {
    const int m0 = k * KCHUNK;
#if USE_TDM
    const int cur = k & 1;
    // Kick off DMA of the next chunk into the other buffer; it overlaps this
    // iteration's WMMA + softmax work (TENSORcnt waited before the barrier).
    if (wave == 0 && k + 1 < NK) {
      const int nxt = cur ^ 1;
      tdm_load_2d((u32)(size_t)&sphi[nxt][0],
                  phib + (size_t)(m0 + KCHUNK) * NCI, NCI, NN, NCI, NCI, KCHUNK);
      tdm_load_2d((u32)(size_t)&sg[nxt][0],
                  gb + (m0 + KCHUNK), NN, NCI, NN, KCHUNK, NCI);
    }
#else
    const int cur = 0;
    __syncthreads();  // previous iteration's readers done
    // Cooperative stage: phi chunk is one contiguous 8KB run; g chunk is 64
    // lines of 128B with stride N.
    for (int i = tid; i < (KCHUNK * NCI) / 8; i += 256) {
      ((v8h*)&sphi[0][0])[i] = ((const v8h*)(phib + (size_t)m0 * NCI))[i];
      const int row = i >> 3, c8 = (i & 7) << 3;
      *(v8h*)&sg[0][row * KCHUNK + c8] =
          *(const v8h*)(gb + (size_t)row * NN + m0 + c8);
    }
    __syncthreads();
#endif

    const _Float16* Pc = &sphi[cur][0];
    const _Float16* Gc = &sg[cur][0];

#pragma unroll
    for (int ms = 0; ms < KCHUNK; ms += 32) {
      // ---- S = theta . phi^T for 32 key columns : 4 WMMAs (LDS B-tiles) ----
      const _Float16* p0 = Pc + (size_t)(ms + lo) * NCI;
      const _Float16* p1 = Pc + (size_t)(ms + 16 + lo) * NCI;
      v8f s0 = {}, s1 = {};
      s0 = wmma_f16(ta0, *(const v16h*)(p0 + hi * 16), s0);
      s0 = wmma_f16(ta1, *(const v16h*)(p0 + 32 + hi * 16), s0);
      s1 = wmma_f16(ta0, *(const v16h*)(p1 + hi * 16), s1);
      s1 = wmma_f16(ta1, *(const v16h*)(p1 + 32 + hi * 16), s1);

      // ---- online softmax update (rows live on 16-lane halves) ----
      float sc[8];
      v8f e0v, e1v;
#pragma unroll
      for (int r = 0; r < 8; ++r) {
        float v = fmaxf(s0[r], s1[r]);
        v = fmaxf(v, __shfl_xor(v, 1));
        v = fmaxf(v, __shfl_xor(v, 2));
        v = fmaxf(v, __shfl_xor(v, 4));
        v = fmaxf(v, __shfl_xor(v, 8));
        const float mn = fmaxf(m[r], v);
        sc[r] = __expf(m[r] - mn);
        m[r] = mn;
        const float e0 = __expf(s0[r] - mn);
        const float e1 = __expf(s1[r] - mn);
        e0v[r] = e0; e1v[r] = e1;
        float s = e0 + e1;
        s += __shfl_xor(s, 1); s += __shfl_xor(s, 2);
        s += __shfl_xor(s, 4); s += __shfl_xor(s, 8);
        l[r] = l[r] * sc[r] + s;
        o0v[r] *= sc[r]; o1v[r] *= sc[r]; o2v[r] *= sc[r]; o3v[r] *= sc[r];
      }

      // ---- stage P (16x32 f16) C-layout -> A-layout via per-wave LDS ----
      // Single-wave producer/consumer: DS ops stay in order within a wave.
      _Float16* pb = &spb[wave][0];
#pragma unroll
      for (int r = 0; r < 8; ++r) {
        const int row = hi * 8 + r;
        pb[row * 32 + lo]      = (_Float16)e0v[r];
        pb[row * 32 + 16 + lo] = (_Float16)e1v[r];
      }
      v16h pa;
      {
        const _Float16* pr = pb + lo * 32;
        v8h q0 = *(const v8h*)(pr + hi * 16);
        v8h q1 = *(const v8h*)(pr + hi * 16 + 8);
#pragma unroll
        for (int j = 0; j < 8; ++j) { pa[j] = q0[j]; pa[8 + j] = q1[j]; }
      }

      // ---- O += P . g  (K=32 chunk, 4 ci-tiles) : 4 WMMAs (LDS B-tiles) ----
      o0v = wmma_f16(pa, *(const v16h*)(Gc + (size_t)(lo)      * KCHUNK + ms + hi * 16), o0v);
      o1v = wmma_f16(pa, *(const v16h*)(Gc + (size_t)(16 + lo) * KCHUNK + ms + hi * 16), o1v);
      o2v = wmma_f16(pa, *(const v16h*)(Gc + (size_t)(32 + lo) * KCHUNK + ms + hi * 16), o2v);
      o3v = wmma_f16(pa, *(const v16h*)(Gc + (size_t)(48 + lo) * KCHUNK + ms + hi * 16), o3v);
    }

#if USE_TDM
    if (wave == 0 && k + 1 < NK) __builtin_amdgcn_s_wait_tensorcnt(0);
    __syncthreads();  // publish next buffer / retire current
#endif
  }

  // ---- normalize and store y[B][N][CI] (f16) ----
#pragma unroll
  for (int r = 0; r < 8; ++r) {
    const float inv = 1.0f / l[r];
    _Float16* yr = y + ((size_t)b * NN + r0 + hi * 8 + r) * NCI;
    yr[lo]      = (_Float16)(o0v[r] * inv);
    yr[16 + lo] = (_Float16)(o1v[r] * inv);
    yr[32 + lo] = (_Float16)(o2v[r] * inv);
    yr[48 + lo] = (_Float16)(o3v[r] * inv);
  }
}

// ---------------------------------------------------------------------------
// Kernel 3: final 1x1 conv (WMMA, K=64) + bias + residual, f32 output.
// ---------------------------------------------------------------------------
__global__ __launch_bounds__(32) void outconv_kernel(
    const _Float16* __restrict__ y,   // [B][N][CI]
    const float* __restrict__ w_w,    // [C][CI]
    const float* __restrict__ w_b,    // [C]
    const float* __restrict__ x,      // [B][C][N]
    float* __restrict__ out)          // [B][C][N]
{
  const int lane = threadIdx.x;
  const int lo = lane & 15, hi = lane >> 4;
  const int c0 = blockIdx.x << 4;
  const int n0 = blockIdx.y << 4;
  const int b  = blockIdx.z;

  const float*    wrow = w_w + (size_t)(c0 + lo) * NCI;
  const _Float16* yrow = y + ((size_t)b * NN + n0 + lo) * NCI;

  v8f acc = {};
#pragma unroll
  for (int kc = 0; kc < NCI; kc += 32) {
    v4f w0 = *(const v4f*)(wrow + kc + hi * 16);
    v4f w1 = *(const v4f*)(wrow + kc + hi * 16 + 4);
    v4f w2 = *(const v4f*)(wrow + kc + hi * 16 + 8);
    v4f w3 = *(const v4f*)(wrow + kc + hi * 16 + 12);
    v16h a;
#pragma unroll
    for (int j = 0; j < 4; ++j) {
      a[j]      = (_Float16)w0[j];
      a[4 + j]  = (_Float16)w1[j];
      a[8 + j]  = (_Float16)w2[j];
      a[12 + j] = (_Float16)w3[j];
    }
    v16h bm = *(const v16h*)(yrow + kc + hi * 16);
    acc = wmma_f16(a, bm, acc);
  }
#pragma unroll
  for (int r = 0; r < 8; ++r) {
    const int c = c0 + hi * 8 + r;
    const size_t idx = ((size_t)b * NC + c) * NN + n0 + lo;
    out[idx] = acc[r] + w_b[c] + x[idx];   // bias + residual fused
  }
}

// ---------------------------------------------------------------------------
extern "C" void kernel_launch(void* const* d_in, const int* in_sizes, int n_in,
                              void* d_out, int out_size, void* d_ws,
                              size_t ws_size, hipStream_t stream) {
  const float* x       = (const float*)d_in[0];
  const float* g_w     = (const float*)d_in[1];
  const float* g_b     = (const float*)d_in[2];
  const float* theta_w = (const float*)d_in[3];
  const float* theta_b = (const float*)d_in[4];
  const float* phi_w   = (const float*)d_in[5];
  const float* phi_b   = (const float*)d_in[6];
  const float* w_w     = (const float*)d_in[7];
  const float* w_b     = (const float*)d_in[8];
  float* out = (float*)d_out;

  // Workspace: 4 x 4MiB f16 arrays (theta, phi, g, y) = 16 MiB total.
  char* ws = (char*)d_ws;
  _Float16* thetaW = (_Float16*)(ws);
  _Float16* phiW   = (_Float16*)(ws + (size_t)(4u << 20));
  _Float16* gW     = (_Float16*)(ws + (size_t)(8u << 20));
  _Float16* yW     = (_Float16*)(ws + (size_t)(12u << 20));

  proj_kernel<<<dim3(12, NN / 16, NB), dim3(32, 1, 1), 0, stream>>>(
      x, theta_w, theta_b, phi_w, phi_b, g_w, g_b, thetaW, phiW, gW);
  attn_kernel<<<dim3(NN / QTILE, NB, 1), dim3(256, 1, 1), 0, stream>>>(
      thetaW, phiW, gW, yW);
  outconv_kernel<<<dim3(NC / 16, NN / 16, NB), dim3(32, 1, 1), 0, stream>>>(
      yW, w_w, w_b, x, out);
}
